// DeformableTransformerEncoderLayer_43748536877316
// MI455X (gfx1250) — matrix operand order repaired and checked by reading, not compiled, not run on hardware
//
#include <hip/hip_runtime.h>

// ---------------------------------------------------------------------------
// Deformable Transformer Encoder Layer for MI455X (gfx1250, wave32, WMMA)
// B=2, S=21760 (128^2+64^2+32^2+16^2), D=256, NH=8, HD=32, NL=4, NP=4, DFF=1024
// ---------------------------------------------------------------------------

#define NTOK   43520          // B * S
#define S_TOT  21760
#define DMODEL 256
#define DFF_   1024

typedef _Float16 half8 __attribute__((ext_vector_type(8)));
typedef _Float16 v16h  __attribute__((ext_vector_type(16)));
typedef float    v8f   __attribute__((ext_vector_type(8)));

union HF16 { v16h v; half8 h[2]; };

// ---------------------------------------------------------------------------
// prep: q_h = f16(src + pos), src_h = f16(src)
// ---------------------------------------------------------------------------
__global__ __launch_bounds__(256) void prep_kernel(const float* __restrict__ src,
                                                   const float* __restrict__ pos,
                                                   _Float16* __restrict__ qh,
                                                   _Float16* __restrict__ srch) {
  size_t idx = (size_t)blockIdx.x * 256 + threadIdx.x;   // grid sized exactly
  float s = src[idx];
  qh[idx]   = (_Float16)(s + pos[idx]);
  srch[idx] = (_Float16)s;
}

// ---------------------------------------------------------------------------
// weight convert + transpose: Wt[m*K + k] = f16(W[k*M + m])
// ---------------------------------------------------------------------------
__global__ __launch_bounds__(256) void wconv_kernel(const float* __restrict__ W,
                                                    _Float16* __restrict__ Wt,
                                                    int K, int M) {
  int idx = blockIdx.x * 256 + threadIdx.x;
  if (idx >= K * M) return;
  int k = idx / M, m = idx % M;
  Wt[(size_t)m * K + k] = (_Float16)W[idx];
}

// ---------------------------------------------------------------------------
// WMMA GEMM:  C[NTOK x M] = A[NTOK x K](f16) * Bt[M x K]^T(f16) + bias
// block = 256 threads = 8 waves; wave computes 32 rows x 64 cols.
// Fragment loads follow the CDNA5 16-bit A(16x32)/B(32x16) VGPR layouts:
//   A elem e: k = (e>=8?16:0) + (lane>=16?8:0) + (e&7)  -> two b128 loads
//   B elem e: k = (e>=8?16:0) + (lane>=16?8:0) + (e&7)  -> two b128 loads
// ---------------------------------------------------------------------------
template <typename OT, bool RELU>
__global__ __launch_bounds__(256) void gemm_kernel(const _Float16* __restrict__ A,
                                                   const _Float16* __restrict__ Bt,
                                                   const float* __restrict__ bias,
                                                   OT* __restrict__ C,
                                                   int K, int M) {
  const int warp = threadIdx.x >> 5;
  const int lane = threadIdx.x & 31;
  const int l16  = lane & 15;
  const int lhi  = lane >> 4;            // 0 or 1

  const int rowBase = blockIdx.x * 256 + warp * 32;
  const int colBase = blockIdx.y * 64;

  v8f acc[2][4];
#pragma unroll
  for (int ri = 0; ri < 2; ++ri)
#pragma unroll
    for (int ci = 0; ci < 4; ++ci) acc[ri][ci] = (v8f)(0.0f);

  for (int k0 = 0; k0 < K; k0 += 32) {
    HF16 afr[2], bfr[4];
#pragma unroll
    for (int ri = 0; ri < 2; ++ri) {
      const _Float16* pa =
          A + (size_t)(rowBase + ri * 16 + l16) * K + k0 + (lhi << 3);
      afr[ri].h[0] = *(const half8*)(pa);
      afr[ri].h[1] = *(const half8*)(pa + 16);
    }
#pragma unroll
    for (int ci = 0; ci < 4; ++ci) {
      const _Float16* pb =
          Bt + (size_t)(colBase + ci * 16 + l16) * K + k0 + (lhi << 3);
      bfr[ci].h[0] = *(const half8*)(pb);
      bfr[ci].h[1] = *(const half8*)(pb + 16);
    }
#pragma unroll
    for (int ri = 0; ri < 2; ++ri)
#pragma unroll
      for (int ci = 0; ci < 4; ++ci)
        acc[ri][ci] = __builtin_amdgcn_wmma_f32_16x16x32_f16(
            false, afr[ri].v, false, bfr[ci].v, (short)0, acc[ri][ci], false,
            false);
  }

  // epilogue: C/D layout -> row = base + r + (lane>=16?8:0), col = base + l16
#pragma unroll
  for (int ri = 0; ri < 2; ++ri)
#pragma unroll
    for (int ci = 0; ci < 4; ++ci) {
      int col = colBase + ci * 16 + l16;
      float bv = bias[col];
#pragma unroll
      for (int r = 0; r < 8; ++r) {
        int row = rowBase + ri * 16 + (lhi << 3) + r;
        float v = acc[ri][ci][r] + bv;
        if (RELU) v = fmaxf(v, 0.0f);
        C[(size_t)row * M + col] = (OT)v;
      }
    }
}

// ---------------------------------------------------------------------------
// Deformable sampling: one wave per (token, head); lane = channel (HD=32).
// softmax over 16 (level,point) logits, bilinear gather from value, f16 out.
// ---------------------------------------------------------------------------
__global__ __launch_bounds__(256) void sample_kernel(const float* __restrict__ value,
                                                     const float* __restrict__ off,
                                                     const float* __restrict__ aw,
                                                     const float* __restrict__ ref,
                                                     _Float16* __restrict__ attnh) {
  const int wid   = blockIdx.x * 8 + (threadIdx.x >> 5);
  const int lane  = threadIdx.x & 31;
  const int token = wid >> 3;
  const int h     = wid & 7;
  const int b     = token / S_TOT;
  const size_t vbase = (size_t)b * S_TOT;

  // softmax over 16 logits (broadcast loads: all lanes same address)
  const float* awp = aw + (size_t)token * 128 + h * 16;
  float lg[16];
  float mx = -1e30f;
#pragma unroll
  for (int j = 0; j < 16; ++j) { lg[j] = awp[j]; mx = fmaxf(mx, lg[j]); }
  float ssum = 0.0f;
#pragma unroll
  for (int j = 0; j < 16; ++j) { lg[j] = __expf(lg[j] - mx); ssum += lg[j]; }
  const float inv = 1.0f / ssum;

  const float* offp = off + (size_t)token * 256 + h * 32;
  const float* refp = ref + (size_t)token * 8;

  constexpr int HLs[4]    = {128, 64, 32, 16};
  constexpr int WLs[4]    = {128, 64, 32, 16};
  constexpr int starts[4] = {0, 16384, 20480, 21504};

  float acc = 0.0f;
#pragma unroll
  for (int l = 0; l < 4; ++l) {
    const int Hl = HLs[l], Wl = WLs[l], st = starts[l];
    const float rx = refp[l * 2 + 0];
    const float ry = refp[l * 2 + 1];
#pragma unroll
    for (int p = 0; p < 4; ++p) {
      const float ox = offp[(l * 4 + p) * 2 + 0];
      const float oy = offp[(l * 4 + p) * 2 + 1];
      // (rx + ox/Wl)*Wl - 0.5 == rx*Wl + ox - 0.5
      const float x = rx * (float)Wl + ox - 0.5f;
      const float y = ry * (float)Hl + oy - 0.5f;
      const float x0f = floorf(x), y0f = floorf(y);
      const float dx = x - x0f, dy = y - y0f;
      const int x0 = (int)x0f, y0 = (int)y0f;
      const float w = lg[l * 4 + p] * inv;

      const float cw[4] = {(1.f - dx) * (1.f - dy) * w, dx * (1.f - dy) * w,
                           (1.f - dx) * dy * w, dx * dy * w};
      const int cx[4] = {x0, x0 + 1, x0, x0 + 1};
      const int cy[4] = {y0, y0, y0 + 1, y0 + 1};
#pragma unroll
      for (int c = 0; c < 4; ++c) {
        const int xi = cx[c], yi = cy[c];
        const bool valid = (xi >= 0) & (xi < Wl) & (yi >= 0) & (yi < Hl);
        const int xc = min(max(xi, 0), Wl - 1);
        const int yc = min(max(yi, 0), Hl - 1);
        const size_t vidx = (size_t)(st + yc * Wl + xc);
        const float vv =
            value[(vbase + vidx) * DMODEL + (size_t)(h * 32 + lane)];
        acc += (valid ? cw[c] : 0.0f) * vv;
      }
    }
  }
  attnh[(size_t)token * DMODEL + h * 32 + lane] = (_Float16)acc;
}

// ---------------------------------------------------------------------------
// fused residual-add + LayerNorm; one wave per row (256 dims, 8 per lane)
// ---------------------------------------------------------------------------
template <bool WRITE_H>
__global__ __launch_bounds__(256) void ln_kernel(const float* __restrict__ a,
                                                 const float* __restrict__ bres,
                                                 const float* __restrict__ g,
                                                 const float* __restrict__ be,
                                                 float* __restrict__ xout,
                                                 _Float16* __restrict__ hout) {
  const int row  = blockIdx.x * 8 + (threadIdx.x >> 5);
  const int lane = threadIdx.x & 31;
  const float* pa = a + (size_t)row * DMODEL;
  const float* pb = bres + (size_t)row * DMODEL;

  float v[8];
  float s = 0.0f;
#pragma unroll
  for (int i = 0; i < 8; ++i) {
    v[i] = pa[lane + 32 * i] + pb[lane + 32 * i];
    s += v[i];
  }
#pragma unroll
  for (int m = 16; m > 0; m >>= 1) s += __shfl_xor(s, m, 32);
  const float mean = s * (1.0f / 256.0f);

  float var = 0.0f;
#pragma unroll
  for (int i = 0; i < 8; ++i) { float d = v[i] - mean; var += d * d; }
#pragma unroll
  for (int m = 16; m > 0; m >>= 1) var += __shfl_xor(var, m, 32);
  const float rinv = rsqrtf(var * (1.0f / 256.0f) + 1e-5f);

#pragma unroll
  for (int i = 0; i < 8; ++i) {
    const int d = lane + 32 * i;
    const float o = (v[i] - mean) * rinv * g[d] + be[d];
    xout[(size_t)row * DMODEL + d] = o;
    if (WRITE_H) hout[(size_t)row * DMODEL + d] = (_Float16)o;
  }
}

// ---------------------------------------------------------------------------
// launcher
// ---------------------------------------------------------------------------
extern "C" void kernel_launch(void* const* d_in, const int* in_sizes, int n_in,
                              void* d_out, int out_size, void* d_ws,
                              size_t ws_size, hipStream_t stream) {
  const float* src   = (const float*)d_in[0];
  const float* pos   = (const float*)d_in[1];
  const float* ref   = (const float*)d_in[2];
  const float* W_off = (const float*)d_in[5];
  const float* b_off = (const float*)d_in[6];
  const float* W_attn= (const float*)d_in[7];
  const float* b_attn= (const float*)d_in[8];
  const float* W_val = (const float*)d_in[9];
  const float* b_val = (const float*)d_in[10];
  const float* W_out = (const float*)d_in[11];
  const float* b_out = (const float*)d_in[12];
  const float* g1    = (const float*)d_in[13];
  const float* be1   = (const float*)d_in[14];
  const float* W_ff1 = (const float*)d_in[15];
  const float* b_ff1 = (const float*)d_in[16];
  const float* W_ff2 = (const float*)d_in[17];
  const float* b_ff2 = (const float*)d_in[18];
  const float* g2    = (const float*)d_in[19];
  const float* be2   = (const float*)d_in[20];

  char* ws = (char*)d_ws;
  size_t o = 0;
  auto alloc = [&](size_t bytes) -> char* {
    char* p = ws + o;
    o += (bytes + 255) & ~(size_t)255;
    return p;
  };
  const size_t NT = NTOK;
  _Float16* qh     = (_Float16*)alloc(NT * DMODEL * 2);
  _Float16* srch   = (_Float16*)alloc(NT * DMODEL * 2);
  float*    value  = (float*)alloc(NT * DMODEL * 4);
  float*    offb   = (float*)alloc(NT * DMODEL * 4);
  float*    awb    = (float*)alloc(NT * 128 * 4);
  float*    attno  = (float*)alloc(NT * DMODEL * 4);
  _Float16* h1     = (_Float16*)alloc(NT * DFF_ * 2);
  _Float16* Wval_t = (_Float16*)alloc(256 * 256 * 2);
  _Float16* Woff_t = (_Float16*)alloc(256 * 256 * 2);
  _Float16* Watt_t = (_Float16*)alloc(128 * 256 * 2);
  _Float16* Wout_t = (_Float16*)alloc(256 * 256 * 2);
  _Float16* Wff1_t = (_Float16*)alloc(1024 * 256 * 2);
  _Float16* Wff2_t = (_Float16*)alloc(256 * 1024 * 2);
  // aliased regions (lifetimes verified against launch order)
  _Float16* attnh = qh;     // q dead after attn-logit GEMM
  float*    xbuf  = offb;   // offsets dead after sampling
  _Float16* xh    = srch;   // src_h dead after value GEMM
  float*    ffn   = value;  // value dead after sampling

  // 1. prep + weight conversion
  prep_kernel<<<NTOK, 256, 0, stream>>>(src, pos, qh, srch);
  wconv_kernel<<<(256 * 256 + 255) / 256, 256, 0, stream>>>(W_val, Wval_t, 256, 256);
  wconv_kernel<<<(256 * 256 + 255) / 256, 256, 0, stream>>>(W_off, Woff_t, 256, 256);
  wconv_kernel<<<(256 * 128 + 255) / 256, 256, 0, stream>>>(W_attn, Watt_t, 256, 128);
  wconv_kernel<<<(256 * 256 + 255) / 256, 256, 0, stream>>>(W_out, Wout_t, 256, 256);
  wconv_kernel<<<(256 * 1024 + 255) / 256, 256, 0, stream>>>(W_ff1, Wff1_t, 256, 1024);
  wconv_kernel<<<(1024 * 256 + 255) / 256, 256, 0, stream>>>(W_ff2, Wff2_t, 1024, 256);

  const int GR = NTOK / 256;  // 170 row-blocks, exact

  // 2. projections
  gemm_kernel<float, false><<<dim3(GR, 4), 256, 0, stream>>>(srch, Wval_t, b_val, value, 256, 256);
  gemm_kernel<float, false><<<dim3(GR, 4), 256, 0, stream>>>(qh, Woff_t, b_off, offb, 256, 256);
  gemm_kernel<float, false><<<dim3(GR, 2), 256, 0, stream>>>(qh, Watt_t, b_attn, awb, 256, 128);

  // 3. deformable sampling (writes attnh, aliases qh)
  sample_kernel<<<NTOK, 256, 0, stream>>>(value, offb, awb, ref, attnh);

  // 4. output projection + LN1
  gemm_kernel<float, false><<<dim3(GR, 4), 256, 0, stream>>>(attnh, Wout_t, b_out, attno, 256, 256);
  ln_kernel<true><<<NTOK / 8, 256, 0, stream>>>(src, attno, g1, be1, xbuf, xh);

  // 5. FFN + LN2
  gemm_kernel<_Float16, true><<<dim3(GR, 16), 256, 0, stream>>>(xh, Wff1_t, b_ff1, h1, 256, 1024);
  gemm_kernel<float, false><<<dim3(GR, 4), 256, 0, stream>>>(h1, Wff2_t, b_ff2, ffn, 1024, 256);
  ln_kernel<false><<<NTOK / 8, 256, 0, stream>>>(xbuf, ffn, g2, be2, (float*)d_out, nullptr);
}
